// DAELoss_12421045420249
// MI455X (gfx1250) — compile-verified, matrix operand order checked by reading, not run on hardware
//
#include <hip/hip_runtime.h>
#include <hip/hip_bf16.h>

#define Bn 512
#define Sn 128
#define Vn 1024

typedef __attribute__((ext_vector_type(16))) _Float16 v16h;
typedef __attribute__((ext_vector_type(8)))  float    v8f;

// ---------------------------------------------------------------------------
// Kernel Z: zero scalar accumulators in workspace
// ---------------------------------------------------------------------------
__global__ void kZ(float* __restrict__ acc, int* __restrict__ ivalid) {
    int i = threadIdx.x;
    if (i < 16) acc[i] = 0.0f;
    if (i == 0) *ivalid = 0;
}

// ---------------------------------------------------------------------------
// Kernel A: per-(b,s) row softmax statistics.
// One 256-thread block (8 wave32s) per row of V=1024 floats.
// Each thread holds 4 contiguous floats in registers (one b128 load),
// so the 256MB tensor is read exactly once.
// Outputs: ce[row] (masked, label-smoothed) and pred[row] (argmax).
// ---------------------------------------------------------------------------
__global__ __launch_bounds__(256) void kA(const float* __restrict__ x,
                                          const int* __restrict__ tgt,
                                          float* __restrict__ ce,
                                          int* __restrict__ pred) {
    const int row  = blockIdx.x;           // 0 .. B*S-1
    const int tid  = threadIdx.x;          // 0 .. 255
    const int lane = tid & 31;
    const int wave = tid >> 5;

    if (tid == 0) __builtin_prefetch(tgt + row, 0, 3);   // global_prefetch_b8

    const float4 xv =
        reinterpret_cast<const float4*>(x + (size_t)row * Vn)[tid];

    // thread-local max + argmax (first occurrence wins)
    float m = xv.x; int mi = 4 * tid;
    if (xv.y > m) { m = xv.y; mi = 4 * tid + 1; }
    if (xv.z > m) { m = xv.z; mi = 4 * tid + 2; }
    if (xv.w > m) { m = xv.w; mi = 4 * tid + 3; }
    float sx = (xv.x + xv.y) + (xv.z + xv.w);

    // wave32 reduce (max, idx)
    #pragma unroll
    for (int off = 16; off; off >>= 1) {
        float ov = __shfl_xor(m, off, 32);
        int   oi = __shfl_xor(mi, off, 32);
        if (ov > m || (ov == m && oi < mi)) { m = ov; mi = oi; }
    }

    __shared__ float wmaxs[8];
    __shared__ int   widxs[8];
    __shared__ float wsx[8], wse[8];
    __shared__ float rowmax_s;
    __shared__ int   rowidx_s;

    if (lane == 0) { wmaxs[wave] = m; widxs[wave] = mi; }
    __syncthreads();

    if (wave == 0) {
        float v = (lane < 8) ? wmaxs[lane] : -3.402823466e38f;
        int   i = (lane < 8) ? widxs[lane] : 0x7fffffff;
        #pragma unroll
        for (int off = 16; off; off >>= 1) {
            float ov = __shfl_xor(v, off, 32);
            int   oi = __shfl_xor(i, off, 32);
            if (ov > v || (ov == v && oi < i)) { v = ov; i = oi; }
        }
        if (lane == 0) { rowmax_s = v; rowidx_s = i; }
    }
    __syncthreads();

    const float rmax = rowmax_s;

    // second pass entirely from registers: sum of exp(x - max)
    float se = __expf(xv.x - rmax) + __expf(xv.y - rmax) +
               __expf(xv.z - rmax) + __expf(xv.w - rmax);

    #pragma unroll
    for (int off = 16; off; off >>= 1) {
        sx += __shfl_xor(sx, off, 32);
        se += __shfl_xor(se, off, 32);
    }
    if (lane == 0) { wsx[wave] = sx; wse[wave] = se; }
    __syncthreads();

    if (tid == 0) {
        float tsx = 0.0f, tse = 0.0f;
        #pragma unroll
        for (int wv = 0; wv < 8; ++wv) { tsx += wsx[wv]; tse += wse[wv]; }
        const float logz = logf(tse);
        const int   t    = tgt[row];
        const float xt   = x[(size_t)row * Vn + t];
        const float nll    = -(xt - rmax - logz);
        const float smooth = -(tsx * (1.0f / (float)Vn) - rmax - logz);
        ce[row]   = (t != 0) ? (0.9f * nll + 0.1f * smooth) : 0.0f;
        pred[row] = rowidx_s;
    }
}

// ---------------------------------------------------------------------------
// Kernel B: per-batch-row statistics. One wave32 per 16 batch rows.
// Phase 1: L[b] = sum_s (t!=0), PL[b] = sum_s (pred!=0) computed as a
//          ones-matmul row-sum on the WMMA unit:
//          D = A(16x32 of 0/1 f16) x B(ones) + D  => D[m][*] = sum_k A[m][k].
//          Counts <=128 are exact (f16 0/1 inputs, f32 accumulate).
// Phase 2: weighted CE, weight sum, correctness, bigram/trigram terms,
//          per-row length stats; wave-shuffle reduce + f32 global atomics.
// ---------------------------------------------------------------------------
__global__ __launch_bounds__(32) void kB(const int* __restrict__ tgt,
                                         const float* __restrict__ ce,
                                         const int* __restrict__ pred,
                                         float* __restrict__ acc,
                                         int* __restrict__ ivalid) {
    const int lane = threadIdx.x;          // 0..31
    const int base = blockIdx.x * 16;      // 16 batch rows per wave
    const int m    = lane & 15;            // my matrix row
    const int hi   = lane >> 4;            // lane half selects K region
    const int row  = base + m;

    const int*   trow = tgt  + row * Sn;
    const int*   prow = pred + row * Sn;
    const float* crow = ce   + row * Sn;

    __builtin_prefetch(trow, 0, 0);
    __builtin_prefetch(prow, 0, 0);

    // ---- Phase 1: WMMA ones-matmul row sums over S=128 (4 chunks of K=32)
    v8f dL = {};
    v8f dP = {};
    v16h onesv;
    #pragma unroll
    for (int e = 0; e < 16; ++e) onesv[e] = (_Float16)1.0f;

    #pragma unroll
    for (int cb = 0; cb < Sn; cb += 32) {
        v16h at, ap;
        #pragma unroll
        for (int e = 0; e < 16; ++e) {
            // 16-bit A 16x32 layout: lanes 0-15 hold K=0..7,16..23;
            // lanes 16-31 hold K=8..15,24..31 (pairs per VGPR).
            int k = ((e < 8) ? e : (8 + e)) + (hi ? 8 : 0);
            int s = cb + k;
            at[e] = (trow[s] != 0) ? (_Float16)1.0f : (_Float16)0.0f;
            ap[e] = (prow[s] != 0) ? (_Float16)1.0f : (_Float16)0.0f;
        }
        dL = __builtin_amdgcn_wmma_f32_16x16x32_f16(
                 false, at, false, onesv, (short)0, dL, false, false);
        dP = __builtin_amdgcn_wmma_f32_16x16x32_f16(
                 false, ap, false, onesv, (short)0, dP, false, false);
    }

    // D layout: VGPR r holds M=r on lanes 0-15, M=8+r on lanes 16-31.
    __shared__ float sL[16], sP[16];
    if (lane == 0) {
        #pragma unroll
        for (int r = 0; r < 8; ++r) { sL[r] = dL[r]; sP[r] = dP[r]; }
    }
    if (lane == 16) {
        #pragma unroll
        for (int r = 0; r < 8; ++r) { sL[8 + r] = dL[r]; sP[8 + r] = dP[r]; }
    }
    __syncthreads();

    const float Lr = sL[m];
    const int   Li = (int)Lr;
    const float Lf = fmaxf(Lr, 1.0f);

    // ---- Phase 2: lanes (l, l+16) split the 128 positions of row m
    float wsum = 0.f, wce = 0.f, corr = 0.f, big = 0.f, tri = 0.f, c126 = 0.f;
    for (int c = 0; c < Sn; c += 4) {
        #pragma unroll
        for (int u = 0; u < 2; ++u) {
            const int s = c + 2 * hi + u;
            const int t = trow[s];
            const int p = prow[s];
            float w = (s < Li) ? (1.0f + (float)s / Lf * 0.5f) : 1.0f;
            if (s == Li - 3 && Li >= 3) w = 3.0f * 0.6f;
            if (s == Li - 2 && Li >= 2) w = 3.0f * 0.8f;
            if (s == Li - 1 && Li >= 1) w = 3.0f;
            wsum += w;
            wce  += w * crow[s];
            if (t != 0) {
                if (p == t) corr += 1.0f;
                if (s < Sn - 2) c126 += 1.0f;
            }
            if (s < Sn - 1) {
                const int t1 = trow[s + 1], p1 = prow[s + 1];
                const bool pe = (p == p1), te = (t == t1), same = (p == t);
                big += (pe ? 1.f : 0.f) + (te ? 1.f : 0.f)
                     - ((pe && te && same) ? 2.f : 0.f);
                if (s < Sn - 2) {
                    const int t2 = trow[s + 2], p2 = prow[s + 2];
                    const bool pe3 = pe && (p1 == p2);
                    const bool te3 = te && (t1 == t2);
                    tri += (pe3 ? 1.f : 0.f) + (te3 ? 1.f : 0.f)
                         - ((pe3 && te3 && same) ? 2.f : 0.f);
                }
            }
        }
    }
    c126 += __shfl_xor(c126, 16, 32);   // lanes 0-15 now hold their row total

    // per-row scalars handled by lanes 0-15 (one row each)
    float absd = 0.f, endok = 0.f, lmatch = 0.f, lsum = 0.f;
    if (lane < 16) {
        const float pl  = sP[lane];
        const float Lrr = sL[lane];
        const int   Lii = (int)Lrr;
        lsum = Lrr;
        absd = fabsf(pl - Lrr);
        int eidx = Lii - 1;
        if (eidx < 0) eidx = 0;
        if (eidx > Sn - 1) eidx = Sn - 1;
        const int r2 = base + lane;
        if (Lii > 0 && pred[r2 * Sn + eidx] == tgt[r2 * Sn + eidx]) endok = 1.0f;
        if (pl == Lrr) lmatch = 1.0f;
        if (c126 > 0.0f) atomicOr(ivalid, 1);
    }

    #pragma unroll
    for (int off = 16; off; off >>= 1) {
        wsum   += __shfl_xor(wsum,   off, 32);
        wce    += __shfl_xor(wce,    off, 32);
        corr   += __shfl_xor(corr,   off, 32);
        big    += __shfl_xor(big,    off, 32);
        tri    += __shfl_xor(tri,    off, 32);
        absd   += __shfl_xor(absd,   off, 32);
        endok  += __shfl_xor(endok,  off, 32);
        lmatch += __shfl_xor(lmatch, off, 32);
        lsum   += __shfl_xor(lsum,   off, 32);
    }
    if (lane == 0) {
        atomicAdd(&acc[0], wce);
        atomicAdd(&acc[1], wsum);
        atomicAdd(&acc[2], corr);
        atomicAdd(&acc[3], lsum);
        atomicAdd(&acc[4], big);
        atomicAdd(&acc[5], tri);
        atomicAdd(&acc[6], absd);
        atomicAdd(&acc[7], endok);
        atomicAdd(&acc[8], lmatch);
    }
}

// ---------------------------------------------------------------------------
// Kernel C: combine accumulators into the four output scalars.
// ---------------------------------------------------------------------------
__global__ void kC(const float* __restrict__ acc,
                   const int* __restrict__ ivalid,
                   float* __restrict__ out) {
    if (threadIdx.x == 0 && blockIdx.x == 0) {
        const float wce   = acc[0], wsum = acc[1], corr = acc[2];
        const float lsum  = acc[3], big  = acc[4], tri  = acc[5];
        const float absd  = acc[6], endk = acc[7], lmt  = acc[8];
        const float weighted = wce / wsum;
        const float lenpen   = 0.1f * (absd / (float)Bn);
        const float bigm = big / (float)(Bn * (Sn - 1) * Vn);
        const float trim = tri / (float)(Bn * (Sn - 2) * Vn);
        const float charng = bigm + ((*ivalid) ? trim : 0.0f);
        out[0] = weighted + lenpen + 0.2f * charng;
        out[1] = (lsum > 0.0f) ? (corr / lsum) : 0.0f;
        out[2] = endk / (float)Bn;
        out[3] = lmt / (float)Bn;
    }
}

// ---------------------------------------------------------------------------
extern "C" void kernel_launch(void* const* d_in, const int* in_sizes, int n_in,
                              void* d_out, int out_size, void* d_ws, size_t ws_size,
                              hipStream_t stream) {
    const float* x   = (const float*)d_in[0];   // (B,S,V) f32
    const int*   tgt = (const int*)d_in[1];     // (B,S)   i32
    float*       out = (float*)d_out;           // 4 f32 scalars

    char*  ws     = (char*)d_ws;
    float* ce     = (float*)ws;                              // B*S f32
    int*   pred   = (int*)(ws + (size_t)Bn * Sn * 4);        // B*S i32
    float* acc    = (float*)(ws + (size_t)Bn * Sn * 8);      // 16 f32
    int*   ivalid = (int*)(acc + 16);

    kZ<<<1, 32, 0, stream>>>(acc, ivalid);
    kA<<<Bn * Sn, 256, 0, stream>>>(x, tgt, ce, pred);
    kB<<<Bn / 16, 32, 0, stream>>>(tgt, ce, pred, acc, ivalid);
    kC<<<1, 1, 0, stream>>>(acc, ivalid, out);
}
